// DeepGRU_1365799600436
// MI455X (gfx1250) — matrix operand order repaired and checked by reading, not compiled, not run on hardware
//
#include <hip/hip_runtime.h>
#include <math.h>

// ---------------------------------------------------------------------------
// DeepGRU for MI455X (gfx1250, wave32, WMMA).
//   BATCH=128, SEQ=512, INPUT=HIDDEN=512, MLP arch [1024,1024].
// Strategy:
//   - convert all weights to bf16, packed in WMMA-B-fragment block layout
//   - precompute Zx/Rx/Hx MLPs over all 65536 rows with a register-blocked
//     (2x4 tiles per wave) WMMA GEMM: 8 wmma per 12 b128 loads
//   - persistent kernel runs the 512-step recurrence with atomic grid barriers
//     (fine-grained tile-per-wave: parallelism-bound, operands L2-resident)
//   - attention GEMM (tanh) + fused softmax/context reduction
// ---------------------------------------------------------------------------

#define GRU_BATCH  128
#define GRU_SEQ    512
#define GRU_IN     512
#define GRU_HID    512
#define GRU_MID    1024
#define GRU_ROWS   (GRU_BATCH * GRU_SEQ)   // 65536
#define GRU_BH     (GRU_BATCH * GRU_HID)   // 65536

typedef unsigned short u16;
typedef unsigned int   u32;
typedef __bf16 bf16_t;
typedef bf16_t v16bf __attribute__((ext_vector_type(16)));
typedef float  v8f   __attribute__((ext_vector_type(8)));

__device__ __forceinline__ u16 f2bf(float f) {
  u32 u = __builtin_bit_cast(u32, f);
  u32 r = 0x7FFFu + ((u >> 16) & 1u);          // round to nearest even
  return (u16)((u + r) >> 16);
}
__device__ __forceinline__ float bf2f(u16 h) {
  u32 u = ((u32)h) << 16;
  return __builtin_bit_cast(float, u);
}

union FragAB { v16bf v; uint4 q[2]; };

#define WMMA_BF16(a, b, c)                                                    \
  __builtin_amdgcn_wmma_f32_16x16x32_bf16(false, (a), false, (b), (short)0,   \
                                          (c), false, false)

// One 16x16 f32 tile = A[16,K] (row-major bf16, lda==K) x Bpacked[K,16].
// Packed B: per (kTile,nTile) 32x16 block, lane l holds 16 contiguous bf16:
//   n = l&15, K = kTile*32 + (l<16?0:16) + e.   (same contiguous k order as the A fragment)
// A fragment: lane l -> row l&15,
//   K = kb..kb+15 contiguous, kb = (l&16)?16:0 (matches packed B).
__device__ __forceinline__ v8f tile_gemm(const u16* __restrict__ A,
                                         const u16* __restrict__ Bp,
                                         int K, int lane) {
  v8f acc = {0.f, 0.f, 0.f, 0.f, 0.f, 0.f, 0.f, 0.f};
  const int row = lane & 15;
  const int kph = (lane & 16) ? 16 : 0;
  const u16* ar = A + row * K + kph;
  const u16* bl = Bp + lane * 16;
  for (int k = 0; k < K; k += 32) {
    FragAB a, b;
    a.q[0] = *(const uint4*)(ar);       // K = kph .. kph+7
    a.q[1] = *(const uint4*)(ar + 8);   // K = kph+8 .. kph+15
    b.q[0] = *(const uint4*)(bl);
    b.q[1] = *(const uint4*)(bl + 8);
    __builtin_prefetch((const void*)(bl + 512), 0, 1);  // next K-block of B
    acc = WMMA_BF16(a.v, b.v, acc);
    ar += 32;
    bl += 512;
  }
  return acc;
}

// C/D layout: VGPR r -> M = r + 8*(lane>=16), N = lane&15.
// act: 0=none 1=relu 2=tanh
__device__ __forceinline__ void tile_store(u16* Cb, float* Cf, int ldc, int lane,
                                           v8f acc, const float* bias16, int act) {
  const int c  = lane & 15;
  const int r0 = (lane & 16) ? 8 : 0;
  const float bv = bias16[c];
#pragma unroll
  for (int r = 0; r < 8; ++r) {
    float v = acc[r] + bv;
    if (act == 1) v = fmaxf(v, 0.0f);
    else if (act == 2) v = tanhf(v);
    if (Cb) Cb[(r0 + r) * ldc + c] = f2bf(v);
    if (Cf) Cf[(r0 + r) * ldc + c] = v;
  }
}

// --------------------------- weight / input prep ---------------------------

__global__ void pack_w_kernel(const float* __restrict__ W, u16* __restrict__ P,
                              int K, int N) {
  const int total = K * N;
  const int kTiles = K >> 5;
  for (int e = blockIdx.x * blockDim.x + threadIdx.x; e < total;
       e += gridDim.x * blockDim.x) {
    int blockI = e >> 9;          // 512 elements per 32x16 block
    int lane   = (e >> 4) & 31;
    int el     = e & 15;
    int nT = blockI / kTiles;     // blocks ordered nTile-major, kTile contiguous
    int kT = blockI - nT * kTiles;
    int k = kT * 32 + ((lane & 16) ? 16 : 0) + el;
    int n = nT * 16 + (lane & 15);
    P[e] = f2bf(W[k * N + n]);
  }
}

__global__ void f32_to_bf16_kernel(const float* __restrict__ src,
                                   u16* __restrict__ dst, int n) {
  for (int i = blockIdx.x * blockDim.x + threadIdx.x; i < n;
       i += gridDim.x * blockDim.x)
    dst[i] = f2bf(src[i]);
}

// ------------------- register-blocked GEMM (2x4 tiles/wave) ----------------
// Each wave computes a 32x64 output block: A fragments reused across 4
// N-tiles, B fragments reused across 2 M-tiles -> 8 WMMA per 12 b128 loads.
// Requires M%32==0, N%64==0.

__global__ void gemm_bf16_blocked(const u16* __restrict__ A, const u16* __restrict__ Bp,
                                  const float* __restrict__ bias, u16* __restrict__ C,
                                  int M, int N, int K, int act) {
  const int lane = threadIdx.x & 31;
  const int wave = blockIdx.x * (blockDim.x >> 5) + (threadIdx.x >> 5);
  const int nw   = gridDim.x * (blockDim.x >> 5);
  const int Nb   = N >> 6;
  const int nblocks = (M >> 5) * Nb;
  const int kb = (K >> 5) * 512;  // ushorts of packed B per N-tile
  const int row = lane & 15;
  const int kph = (lane & 16) ? 16 : 0;

  for (int blk = wave; blk < nblocks; blk += nw) {
    int mb = blk / Nb, nb = blk - mb * Nb;
    const u16* ar0 = A + (mb * 32 + row) * K + kph;
    const u16* ar1 = ar0 + 16 * K;
    const u16* bl  = Bp + (nb * 4) * kb + lane * 16;

    v8f acc[2][4];
#pragma unroll
    for (int i = 0; i < 2; ++i)
#pragma unroll
      for (int j = 0; j < 4; ++j)
        acc[i][j] = v8f{0.f, 0.f, 0.f, 0.f, 0.f, 0.f, 0.f, 0.f};

    for (int k = 0; k < K; k += 32) {
      FragAB a0, a1, b[4];
      a0.q[0] = *(const uint4*)(ar0);
      a0.q[1] = *(const uint4*)(ar0 + 8);
      a1.q[0] = *(const uint4*)(ar1);
      a1.q[1] = *(const uint4*)(ar1 + 8);
#pragma unroll
      for (int j = 0; j < 4; ++j) {
        const u16* bj = bl + j * kb;
        b[j].q[0] = *(const uint4*)(bj);
        b[j].q[1] = *(const uint4*)(bj + 8);
        __builtin_prefetch((const void*)(bj + 512), 0, 1);
      }
#pragma unroll
      for (int j = 0; j < 4; ++j) {
        acc[0][j] = WMMA_BF16(a0.v, b[j].v, acc[0][j]);
        acc[1][j] = WMMA_BF16(a1.v, b[j].v, acc[1][j]);
      }
      ar0 += 32;
      ar1 += 32;
      bl += 512;
    }

#pragma unroll
    for (int i = 0; i < 2; ++i)
#pragma unroll
      for (int j = 0; j < 4; ++j)
        tile_store(C + (mb * 32 + i * 16) * N + nb * 64 + j * 16, nullptr, N,
                   lane, acc[i][j], bias + nb * 64 + j * 16, act);
  }
}

// ------------------------- persistent recurrent scan -----------------------

struct GruArgs {
  const u16 *W1z, *W2z, *W3z, *W1r, *W2r, *W3r, *W1h, *W2h, *W3h;
  const float *b1z, *b2z, *b3z, *b1r, *b2r, *b3r, *b1h, *b2h, *b3h;
  const u16 *Xz, *Xr, *Xh;        // precomputed x-side gate terms [B,S,H] bf16
  float *hn; u16 *hnb;            // hidden state fp32 + bf16 mirror [B,H]
  u16 *a1z, *a1r, *a1h;           // [128,1024] bf16
  u16 *a2z, *a2r, *a2h;           // [128,1024] bf16
  float *pz, *pr, *ph, *Zbuf;     // [128,512] fp32
  u16 *ubuf, *hs;                 // hn*R bf16; hs [T,B,H] bf16
  u32 *barCnt, *barGen;
};

__device__ void grid_barrier(u32* cnt, u32* gen, u32 nwg) {
  __syncthreads();
  if (threadIdx.x == 0) {
    __threadfence();
    volatile u32* vg = (volatile u32*)gen;
    u32 g = *vg;
    if (atomicAdd(cnt, 1u) == nwg - 1u) {
      *cnt = 0u;
      __threadfence();
      atomicAdd(gen, 1u);
    } else {
      while (*vg == g) __builtin_amdgcn_s_sleep(2);
      __threadfence();
    }
  }
  __syncthreads();
}

__global__ void gru_scan_kernel(GruArgs A) {
  const int lane = threadIdx.x & 31;
  const int gw = blockIdx.x * (blockDim.x >> 5) + (threadIdx.x >> 5);
  const int nw = gridDim.x * (blockDim.x >> 5);
  const int gt = blockIdx.x * blockDim.x + threadIdx.x;
  const int nt = gridDim.x * blockDim.x;
  const u32 nwg = gridDim.x;

  for (int t = 0; t < GRU_SEQ; ++t) {
    // ---- Stage A: a1{z,r} = relu(hnb @ W1 + b1)  [128,1024], K=512 ----
    for (int w = gw; w < 2 * 8 * 64; w += nw) {
      int mat = w >> 9, ti = w & 511;
      int mt = ti >> 6, nti = ti & 63;
      const u16* W = mat ? A.W1r : A.W1z;
      const float* bs = mat ? A.b1r : A.b1z;
      u16* dst = mat ? A.a1r : A.a1z;
      v8f acc = tile_gemm(A.hnb + mt * 16 * 512, W + nti * 16 * 512, 512, lane);
      tile_store(dst + mt * 16 * 1024 + nti * 16, nullptr, 1024, lane, acc,
                 bs + nti * 16, 1);
    }
    grid_barrier(A.barCnt, A.barGen, nwg);

    // ---- Stage B: a2 = relu(a1 @ W2 + b2)  [128,1024], K=1024 ----
    for (int w = gw; w < 2 * 8 * 64; w += nw) {
      int mat = w >> 9, ti = w & 511;
      int mt = ti >> 6, nti = ti & 63;
      const u16* W = mat ? A.W2r : A.W2z;
      const float* bs = mat ? A.b2r : A.b2z;
      const u16* src = mat ? A.a1r : A.a1z;
      u16* dst = mat ? A.a2r : A.a2z;
      v8f acc = tile_gemm(src + mt * 16 * 1024, W + nti * 32 * 512, 1024, lane);
      tile_store(dst + mt * 16 * 1024 + nti * 16, nullptr, 1024, lane, acc,
                 bs + nti * 16, 1);
    }
    grid_barrier(A.barCnt, A.barGen, nwg);

    // ---- Stage C: p{z,r} = a2 @ W3 + b3  [128,512] fp32, K=1024 ----
    for (int w = gw; w < 2 * 8 * 32; w += nw) {
      int mat = w >> 8, ti = w & 255;
      int mt = ti >> 5, nti = ti & 31;
      const u16* W = mat ? A.W3r : A.W3z;
      const float* bs = mat ? A.b3r : A.b3z;
      float* dst = mat ? A.pr : A.pz;
      const u16* src = mat ? A.a2r : A.a2z;
      v8f acc = tile_gemm(src + mt * 16 * 1024, W + nti * 32 * 512, 1024, lane);
      tile_store(nullptr, dst + mt * 16 * 512 + nti * 16, 512, lane, acc,
                 bs + nti * 16, 0);
    }
    grid_barrier(A.barCnt, A.barGen, nwg);

    // ---- Stage D: Z, R gates + u = hn*R ----
    for (int i = gt; i < GRU_BH; i += nt) {
      int b = i >> 9, hcol = i & 511;
      int xi = (b * GRU_SEQ + t) * GRU_HID + hcol;
      float z = 1.f / (1.f + __expf(-(A.pz[i] + bf2f(A.Xz[xi]))));
      float r = 1.f / (1.f + __expf(-(A.pr[i] + bf2f(A.Xr[xi]))));
      A.Zbuf[i] = z;
      A.ubuf[i] = f2bf(A.hn[i] * r);
    }
    grid_barrier(A.barCnt, A.barGen, nwg);

    // ---- Stage E: a1h = relu(u @ W1h + b1h)  K=512 ----
    for (int w = gw; w < 8 * 64; w += nw) {
      int mt = w >> 6, nti = w & 63;
      v8f acc = tile_gemm(A.ubuf + mt * 16 * 512, A.W1h + nti * 16 * 512, 512, lane);
      tile_store(A.a1h + mt * 16 * 1024 + nti * 16, nullptr, 1024, lane, acc,
                 A.b1h + nti * 16, 1);
    }
    grid_barrier(A.barCnt, A.barGen, nwg);

    // ---- Stage F: a2h = relu(a1h @ W2h + b2h)  K=1024 ----
    for (int w = gw; w < 8 * 64; w += nw) {
      int mt = w >> 6, nti = w & 63;
      v8f acc = tile_gemm(A.a1h + mt * 16 * 1024, A.W2h + nti * 32 * 512, 1024, lane);
      tile_store(A.a2h + mt * 16 * 1024 + nti * 16, nullptr, 1024, lane, acc,
                 A.b2h + nti * 16, 1);
    }
    grid_barrier(A.barCnt, A.barGen, nwg);

    // ---- Stage G: ph = a2h @ W3h + b3h  fp32, K=1024 ----
    for (int w = gw; w < 8 * 32; w += nw) {
      int mt = w >> 5, nti = w & 31;
      v8f acc = tile_gemm(A.a2h + mt * 16 * 1024, A.W3h + nti * 32 * 512, 1024, lane);
      tile_store(nullptr, A.ph + mt * 16 * 512 + nti * 16, 512, lane, acc,
                 A.b3h + nti * 16, 0);
    }
    grid_barrier(A.barCnt, A.barGen, nwg);

    // ---- Stage H: hn = hn*Z + (1-Z)*tanh(ph + Xh); emit hs[t] ----
    for (int i = gt; i < GRU_BH; i += nt) {
      int b = i >> 9, hcol = i & 511;
      int xi = (b * GRU_SEQ + t) * GRU_HID + hcol;
      float hh = tanhf(A.ph[i] + bf2f(A.Xh[xi]));
      float z = A.Zbuf[i];
      float v = A.hn[i] * z + (1.f - z) * hh;
      A.hn[i] = v;
      u16 hb = f2bf(v);
      A.hnb[i] = hb;
      A.hs[t * GRU_BH + i] = hb;
    }
    grid_barrier(A.barCnt, A.barGen, nwg);
  }
}

// ------------------------ attention softmax + context ----------------------

__global__ void attn_context_kernel(const u16* __restrict__ aw,
                                    const u16* __restrict__ hs,
                                    float* __restrict__ out) {
  int i = blockIdx.x * blockDim.x + threadIdx.x;  // (b,h) pair
  if (i >= GRU_BH) return;
  float mx = -3.0e38f;
  for (int t = 0; t < GRU_SEQ; ++t)
    mx = fmaxf(mx, bf2f(aw[t * GRU_BH + i]));
  float s = 0.f, ws = 0.f;
  for (int t = 0; t < GRU_SEQ; ++t) {
    float e = __expf(bf2f(aw[t * GRU_BH + i]) - mx);
    s += e;
    ws += e * bf2f(hs[t * GRU_BH + i]);
  }
  out[i] = ws / s;
}

// --------------------------------- launch ----------------------------------

extern "C" void kernel_launch(void* const* d_in, const int* in_sizes, int n_in,
                              void* d_out, int out_size, void* d_ws, size_t ws_size,
                              hipStream_t stream) {
  (void)in_sizes; (void)n_in; (void)out_size; (void)ws_size;

  // Input order (setup_inputs dict order): x, then for each MLP in
  // {Zh, Zx, Rh, Rx, Hh, Hx}: W1,W2,W3,b1,b2,b3; then att W, att b.
  const float* x = (const float*)d_in[0];
  auto mlpW = [&](int m, int l) { return (const float*)d_in[1 + m * 6 + l]; };
  auto mlpB = [&](int m, int l) { return (const float*)d_in[1 + m * 6 + 3 + l]; };
  const float* attW = (const float*)d_in[37];
  const float* attB = (const float*)d_in[38];

  // Workspace layout.
  size_t off = 0;
  auto alloc = [&](size_t bytes) -> void* {
    off = (off + 255) & ~(size_t)255;
    void* p = (char*)d_ws + off;
    off += bytes;
    return p;
  };
  u16* Wp[6][3];
  for (int m = 0; m < 6; ++m) {
    Wp[m][0] = (u16*)alloc((size_t)512 * 1024 * 2);
    Wp[m][1] = (u16*)alloc((size_t)1024 * 1024 * 2);
    Wp[m][2] = (u16*)alloc((size_t)1024 * 512 * 2);
  }
  u16* attWp = (u16*)alloc((size_t)512 * 512 * 2);
  u16* xb = (u16*)alloc((size_t)GRU_ROWS * GRU_IN * 2);
  u16* h1 = (u16*)alloc((size_t)GRU_ROWS * GRU_MID * 2);
  u16* h2 = (u16*)alloc((size_t)GRU_ROWS * GRU_MID * 2);
  u16* Xz = (u16*)alloc((size_t)GRU_ROWS * GRU_HID * 2);
  u16* Xr = (u16*)alloc((size_t)GRU_ROWS * GRU_HID * 2);
  u16* Xh = (u16*)alloc((size_t)GRU_ROWS * GRU_HID * 2);
  u16* hs = (u16*)alloc((size_t)GRU_SEQ * GRU_BH * 2);
  u16* aw = (u16*)alloc((size_t)GRU_SEQ * GRU_BH * 2);
  float* hn = (float*)alloc((size_t)GRU_BH * 4);
  u16* hnb = (u16*)alloc((size_t)GRU_BH * 2);
  u16* a1z = (u16*)alloc((size_t)128 * 1024 * 2);
  u16* a1r = (u16*)alloc((size_t)128 * 1024 * 2);
  u16* a1h = (u16*)alloc((size_t)128 * 1024 * 2);
  u16* a2z = (u16*)alloc((size_t)128 * 1024 * 2);
  u16* a2r = (u16*)alloc((size_t)128 * 1024 * 2);
  u16* a2h = (u16*)alloc((size_t)128 * 1024 * 2);
  float* pz = (float*)alloc((size_t)GRU_BH * 4);
  float* pr = (float*)alloc((size_t)GRU_BH * 4);
  float* ph = (float*)alloc((size_t)GRU_BH * 4);
  float* Zb = (float*)alloc((size_t)GRU_BH * 4);
  u16* ub = (u16*)alloc((size_t)GRU_BH * 2);
  u32* bar = (u32*)alloc(256);

  // 1) Pack all weights into bf16 WMMA-fragment layout (L2-resident, 26 MB).
  for (int m = 0; m < 6; ++m) {
    pack_w_kernel<<<512, 256, 0, stream>>>(mlpW(m, 0), Wp[m][0], 512, 1024);
    pack_w_kernel<<<512, 256, 0, stream>>>(mlpW(m, 1), Wp[m][1], 1024, 1024);
    pack_w_kernel<<<512, 256, 0, stream>>>(mlpW(m, 2), Wp[m][2], 1024, 512);
  }
  pack_w_kernel<<<512, 256, 0, stream>>>(attW, attWp, 512, 512);
  f32_to_bf16_kernel<<<4096, 256, 0, stream>>>(x, xb, GRU_ROWS * GRU_IN);

  // 2) Precompute x-side gate MLPs (parallel: half of all FLOPs).
  //    m=1 (Zx) -> Xz, m=3 (Rx) -> Xr, m=5 (Hx) -> Xh.
  const int xm[3] = {1, 3, 5};
  u16* xd[3] = {Xz, Xr, Xh};
  for (int j = 0; j < 3; ++j) {
    int m = xm[j];
    gemm_bf16_blocked<<<2048, 256, 0, stream>>>(xb, Wp[m][0], mlpB(m, 0), h1,
                                                GRU_ROWS, GRU_MID, GRU_IN, 1);
    gemm_bf16_blocked<<<2048, 256, 0, stream>>>(h1, Wp[m][1], mlpB(m, 1), h2,
                                                GRU_ROWS, GRU_MID, GRU_MID, 1);
    gemm_bf16_blocked<<<2048, 256, 0, stream>>>(h2, Wp[m][2], mlpB(m, 2), xd[j],
                                                GRU_ROWS, GRU_HID, GRU_MID, 0);
  }

  // 3) Recurrent scan: persistent kernel, 64 WGs x 8 wave32, grid barriers.
  hipMemsetAsync(hn, 0, (size_t)GRU_BH * 4, stream);
  hipMemsetAsync(hnb, 0, (size_t)GRU_BH * 2, stream);
  hipMemsetAsync(bar, 0, 256, stream);
  GruArgs ga;
  ga.W1z = Wp[0][0]; ga.W2z = Wp[0][1]; ga.W3z = Wp[0][2];
  ga.W1r = Wp[2][0]; ga.W2r = Wp[2][1]; ga.W3r = Wp[2][2];
  ga.W1h = Wp[4][0]; ga.W2h = Wp[4][1]; ga.W3h = Wp[4][2];
  ga.b1z = mlpB(0, 0); ga.b2z = mlpB(0, 1); ga.b3z = mlpB(0, 2);
  ga.b1r = mlpB(2, 0); ga.b2r = mlpB(2, 1); ga.b3r = mlpB(2, 2);
  ga.b1h = mlpB(4, 0); ga.b2h = mlpB(4, 1); ga.b3h = mlpB(4, 2);
  ga.Xz = Xz; ga.Xr = Xr; ga.Xh = Xh;
  ga.hn = hn; ga.hnb = hnb;
  ga.a1z = a1z; ga.a1r = a1r; ga.a1h = a1h;
  ga.a2z = a2z; ga.a2r = a2r; ga.a2h = a2h;
  ga.pz = pz; ga.pr = pr; ga.ph = ph; ga.Zbuf = Zb;
  ga.ubuf = ub; ga.hs = hs;
  ga.barCnt = bar; ga.barGen = bar + 1;
  gru_scan_kernel<<<64, 256, 0, stream>>>(ga);

  // 4) Attention logits: aw = tanh(hs @ attW + attB)  [T*B, 512].
  gemm_bf16_blocked<<<2048, 256, 0, stream>>>(hs, attWp, attB, aw,
                                              GRU_ROWS, GRU_HID, GRU_HID, 2);

  // 5) Softmax over T + context reduction -> out [B, H] fp32.
  attn_context_kernel<<<256, 256, 0, stream>>>(aw, hs, (float*)d_out);
}
